// Graph_Generator_54606214201431
// MI455X (gfx1250) — matrix operand order repaired and checked by reading, not compiled, not run on hardware
//
#include <hip/hip_runtime.h>

// ---------------------------------------------------------------------------
// Shapes (fixed by the reference)
// ---------------------------------------------------------------------------
#define B_    64
#define HID_  128
#define NODE_ 512
#define L_    12
#define S0_   32
#define S1_   32

typedef __attribute__((ext_vector_type(16))) __bf16 v16bf;
typedef __attribute__((ext_vector_type(8)))  __bf16 v8bf;
typedef __attribute__((ext_vector_type(8)))  float  v8f;
typedef __attribute__((ext_vector_type(4)))  int    v4i_t;

#define ADP2_BYTES ((size_t)B_ * NODE_ * L_ * sizeof(float))   // 1.5 MB

// LDS staging of E_b: 512 rows x 128 bf16, padded to 136 elems (272 B) per row
// so fragment loads (row stride = 68 DWORDs -> 4 banks/lane) are conflict-free.
#define PITCH_     136
#define LDS_BYTES  ((size_t)NODE_ * PITCH_ * sizeof(__bf16))   // 139,264 B

#define AS1 __attribute__((address_space(1)))
#define AS3 __attribute__((address_space(3)))

#if defined(__has_builtin)
#  if __has_builtin(__builtin_amdgcn_global_load_async_to_lds_b128)
#    define HAVE_ASYNC_LDS 1
#  endif
#endif

__device__ __forceinline__ void wait_async_zero() {
#if defined(__has_builtin) && __has_builtin(__builtin_amdgcn_s_wait_asynccnt)
    __builtin_amdgcn_s_wait_asynccnt(0);
#else
    asm volatile("s_wait_asynccnt 0x0" ::: "memory");
#endif
}

// ---------------------------------------------------------------------------
// Kernel 1: adp2[b][v][l] = softmax_l( weight_n @ (weight_t[t_i[b]] @ k) )
// Tiny: one block per batch.
// ---------------------------------------------------------------------------
__global__ void k_adp2(const int* __restrict__ t_i,
                       const float* __restrict__ weight_t,
                       const float* __restrict__ weight_n,
                       const float* __restrict__ kmat,
                       float* __restrict__ adp2) {
    __shared__ float s_wt[S0_];
    __shared__ float s_adp1[S1_][L_];

    const int b   = blockIdx.x;
    const int tid = threadIdx.x;

    if (tid < S0_) s_wt[tid] = weight_t[t_i[b] * S0_ + tid];
    __syncthreads();

    for (int idx = tid; idx < S1_ * L_; idx += blockDim.x) {
        const int c = idx / L_, d = idx % L_;
        float s = 0.f;
        #pragma unroll
        for (int q = 0; q < S0_; ++q)
            s += s_wt[q] * kmat[(q * S1_ + c) * L_ + d];
        s_adp1[c][d] = s;
    }
    __syncthreads();

    for (int e = tid; e < NODE_; e += blockDim.x) {
        float v[L_];
        #pragma unroll
        for (int d = 0; d < L_; ++d) v[d] = 0.f;
        for (int c = 0; c < S1_; ++c) {
            const float wn = weight_n[e * S1_ + c];
            #pragma unroll
            for (int d = 0; d < L_; ++d) v[d] += wn * s_adp1[c][d];
        }
        float mx = v[0];
        #pragma unroll
        for (int d = 1; d < L_; ++d) mx = fmaxf(mx, v[d]);
        float sum = 0.f;
        #pragma unroll
        for (int d = 0; d < L_; ++d) { v[d] = __expf(v[d] - mx); sum += v[d]; }
        const float inv = 1.f / sum;
        float* o = adp2 + ((size_t)b * NODE_ + e) * L_;
        #pragma unroll
        for (int d = 0; d < L_; ++d) o[d] = v[d] * inv;
    }
}

// ---------------------------------------------------------------------------
// Kernel 2: E[b][v][c] = bf16( sum_l (x[b][c][v][l] + n_emb[c][v][l]) *
//                              adp2[b][v][l] )
// Streams x (201 MB) exactly once; consecutive lanes walk v so the 48B/thread
// reads of x and n_emb are fully coalesced b128 loads.
// ---------------------------------------------------------------------------
__global__ void k_en(const float* __restrict__ x,
                     const float* __restrict__ n_emb,
                     const float* __restrict__ adp2,
                     __bf16* __restrict__ E) {
    const int b   = blockIdx.y;
    const int idx = blockIdx.x * blockDim.x + threadIdx.x;  // 0..HID_*NODE_-1
    const int v   = idx & (NODE_ - 1);
    const int c   = idx >> 9;

    const float4* xp = (const float4*)(x + ((((size_t)b * HID_ + c) * NODE_ + v) * L_));
    const float4* np = (const float4*)(n_emb + (((size_t)c * NODE_ + v) * L_));
    const float*  w  = adp2 + ((size_t)b * NODE_ + v) * L_;

    float s = 0.f;
    #pragma unroll
    for (int q = 0; q < 3; ++q) {
        const float4 xv = xp[q];
        const float4 nv = np[q];
        s += (xv.x + nv.x) * w[4 * q + 0];
        s += (xv.y + nv.y) * w[4 * q + 1];
        s += (xv.z + nv.z) * w[4 * q + 2];
        s += (xv.w + nv.w) * w[4 * q + 3];
    }
    E[((size_t)b * NODE_ + v) * HID_ + c] = (__bf16)s;
}

// ---------------------------------------------------------------------------
// Kernel 3: out[b] = softmax_row( relu( (E @ E^T) / sqrt(HID) ) )
//
// Stage 1: async-DMA the whole per-batch E (512x128 bf16 = 128 KB) into LDS
//          (GLOBAL_LOAD_ASYNC_TO_LDS_B128, ASYNCcnt), row pitch padded to
//          272 B for bank-conflict-free fragment reads.
// Stage 2: one wave per 16-row strip: 32 column fragments * 4 K-steps
//          -> 128 x v_wmma_f32_16x16x32_bf16 per wave, fed by ds_load_b128.
// Stage 3: relu/scale/row-softmax fused in-register (row m of a C/D fragment
//          lives at VGPR index r across one 16-lane half -> shfl_xor 1..8).
//
// bf16 A frag (16x32), lane L:  row m0+(L&15), K = k0+8*(L>=16)+{0..7,16..23}
// bf16 B frag (32x16), lane L:  col n0+(L&15), K = k0+8*(L>=16)+{0..7,16..23}
//   (B[k][n] = E[n][k], so both frags are contiguous runs of row-major E)
// ---------------------------------------------------------------------------
__global__ void __launch_bounds__(256, 1)
k_gram_softmax(const __bf16* __restrict__ E, float* __restrict__ out) {
    extern __shared__ __bf16 sE[];                 // [NODE_][PITCH_]

    const int b    = blockIdx.y;
    const int tid  = threadIdx.x;
    const int lane = tid & 31;
    const int wv   = tid >> 5;
    const int m0   = (blockIdx.x * 8 + wv) * 16;   // strip base row
    const int half = lane >> 4;                    // 0 | 1
    const int l15  = lane & 15;

    const __bf16* Eb = E + (size_t)b * NODE_ * HID_;

    // ---- Stage E_b into LDS (8192 chunks of 16 B; 32 per thread) ----------
#if defined(HAVE_ASYNC_LDS)
    {
        #pragma unroll
        for (int i = 0; i < 32; ++i) {
            const int n   = tid + i * 256;         // 16B-chunk index
            const int row = n >> 4;                // 16 chunks per 256B row
            const int cc  = n & 15;
            AS1 v4i_t* gsrc =
                (AS1 v4i_t*)((const char*)Eb + (size_t)n * 16);
            AS3 v4i_t* ldst =
                (AS3 v4i_t*)((char*)sE + row * (PITCH_ * 2) + cc * 16);
            __builtin_amdgcn_global_load_async_to_lds_b128(gsrc, ldst, 0, 0);
        }
        wait_async_zero();
    }
#else
    {
        for (int n = tid; n < NODE_ * (HID_ / 8); n += 256) {
            const int row = n >> 4, cc = n & 15;
            *(v8bf*)(sE + row * PITCH_ + cc * 8) =
                *(const v8bf*)(Eb + (size_t)n * 8);
        }
    }
#endif
    __syncthreads();

    // ---- WMMA gram over K = HID_ ------------------------------------------
    v8f acc[32];
    #pragma unroll
    for (int j = 0; j < 32; ++j) acc[j] = (v8f)0.f;

    #pragma unroll
    for (int k0 = 0; k0 < HID_; k0 += 32) {
        const __bf16* pa = sE + (m0 + l15) * PITCH_ + k0 + half * 8;
        v16bf a;
        *((v8bf*)&a)     = *(const v8bf*)pa;          // K +0..7
        *((v8bf*)&a + 1) = *(const v8bf*)(pa + 16);   // K +16..23

        #pragma unroll
        for (int j = 0; j < 32; ++j) {
            const __bf16* pb = sE + (j * 16 + l15) * PITCH_ + k0 + half * 8;
            v16bf bb;
            *((v8bf*)&bb)     = *(const v8bf*)pb;        // K +0..7
            *((v8bf*)&bb + 1) = *(const v8bf*)(pb + 16); // K +16..23
            acc[j] = __builtin_amdgcn_wmma_f32_16x16x32_bf16(
                false, a, false, bb, (short)0, acc[j], false, false);
        }
    }

    // ---- Fused scale + relu + row softmax ---------------------------------
    const float norm = 0.08838834764831845f;       // 1/sqrt(128)
    #pragma unroll
    for (int r = 0; r < 8; ++r) {
        float vr[32];
        float mx = 0.f;                            // relu floor
        #pragma unroll
        for (int j = 0; j < 32; ++j) {
            float t = fmaxf(acc[j][r] * norm, 0.f);
            vr[j] = t;
            mx = fmaxf(mx, t);
        }
        #pragma unroll
        for (int m = 1; m <= 8; m <<= 1) mx = fmaxf(mx, __shfl_xor(mx, m, 32));

        float sum = 0.f;
        #pragma unroll
        for (int j = 0; j < 32; ++j) { vr[j] = __expf(vr[j] - mx); sum += vr[j]; }
        #pragma unroll
        for (int m = 1; m <= 8; m <<= 1) sum += __shfl_xor(sum, m, 32);

        const float inv = 1.f / sum;
        const int row = m0 + half * 8 + r;
        float* orow = out + ((size_t)b * NODE_ + row) * NODE_;
        #pragma unroll
        for (int j = 0; j < 32; ++j) orow[j * 16 + l15] = vr[j] * inv;
    }
}

// ---------------------------------------------------------------------------
// Launch
// ---------------------------------------------------------------------------
extern "C" void kernel_launch(void* const* d_in, const int* in_sizes, int n_in,
                              void* d_out, int out_size, void* d_ws, size_t ws_size,
                              hipStream_t stream) {
    const float* x        = (const float*)d_in[0];
    const int*   t_i      = (const int*)  d_in[1];
    const float* weight_t = (const float*)d_in[2];
    const float* weight_n = (const float*)d_in[3];
    const float* kmat     = (const float*)d_in[4];
    const float* n_emb    = (const float*)d_in[5];
    float*       out      = (float*)d_out;

    float*  adp2 = (float*)d_ws;
    __bf16* E    = (__bf16*)((char*)d_ws + ADP2_BYTES);

    // 1) time/node mixing weights + softmax over L   (tiny)
    k_adp2<<<dim3(B_), dim3(256), 0, stream>>>(t_i, weight_t, weight_n, kmat, adp2);

    // 2) e_n reduction over L, fused x+n_emb, bf16 transpose-store (201 MB stream)
    k_en<<<dim3((HID_ * NODE_) / 256, B_), dim3(256), 0, stream>>>(x, n_emb, adp2, E);

    // 3) async-LDS staged WMMA gram + fused relu/scale/row-softmax
    k_gram_softmax<<<dim3(4, B_), dim3(256), LDS_BYTES, stream>>>(E, out);
}
